// CloseScannedRNN_36661840838845
// MI455X (gfx1250) — matrix operand order repaired and checked by reading, not compiled, not run on hardware
//
#include <hip/hip_runtime.h>
#include <hip/hip_bf16.h>

// GRU scan for MI455X (gfx1250), wave32 + WMMA bf16 + async LDS staging +
// single persistent scan kernel with device-wide barrier.
//
//  1) Convert x -> bf16, transpose Wi/Wh -> [3H x K] bf16 (one-time).
//  2) GI = X @ Wi + bi as one big WMMA GEMM (no sequential dependence).
//  3) ONE persistent kernel runs all 512 recurrent steps: each of the 32
//     blocks async-stages its 48KB Wh column slice into LDS once
//     (global_load_async_to_lds_b128 + s_wait_asynccnt), then per step runs
//     the K-loop as ds_load + v_wmma with register-double-buffered A
//     fragments, fused gate epilogue, and an atomic grid barrier between
//     steps (h ping-pongs between two bf16 buffers).

typedef __attribute__((ext_vector_type(16))) __bf16 v16bf;
typedef __attribute__((ext_vector_type(8)))  __bf16 v8bf;
typedef __attribute__((ext_vector_type(8)))  float  v8f;

#define T_STEPS 512
#define BATCH   64
#define DIM     512
#define HID     512
#define H3      1536
#define NBLK    32   // HID / 16 column slices

__device__ __forceinline__ float sigmoidf_(float x) {
  return 1.0f / (1.0f + __expf(-x));
}
__device__ __forceinline__ float tanh_fast(float x) {
  float e = __expf(-2.0f * x);
  return (1.0f - e) / (1.0f + e);
}

__device__ __forceinline__ v16bf combine_frag(v8bf lo, v8bf hi) {
  v16bf a;
#pragma unroll
  for (int i = 0; i < 8; ++i) { a[i] = lo[i]; a[i + 8] = hi[i]; }
  return a;
}

// Issue one 16-byte async global->LDS copy (per-lane addresses).
__device__ __forceinline__ void async_copy_b128(const __bf16* gsrc, const __bf16* lds_dst) {
  unsigned lds_off = (unsigned)(unsigned long long)lds_dst;   // flat LDS addr low 32 = LDS offset
  unsigned long long ga = (unsigned long long)gsrc;
  asm volatile("global_load_async_to_lds_b128 %0, %1, off"
               :: "v"(lds_off), "v"(ga) : "memory");
}
__device__ __forceinline__ void wait_async_zero() {
  asm volatile("s_wait_asynccnt 0x0" ::: "memory");
}

// ---------------- conversion / transpose (one-time) ----------------

__global__ void k_f32_to_bf16(const float* __restrict__ src,
                              __bf16* __restrict__ dst, int n) {
  int i = blockIdx.x * blockDim.x + threadIdx.x;
  if (i < n) dst[i] = (__bf16)src[i];
}

// W [K x N] f32 row-major  ->  WT [N x K] bf16 row-major
__global__ void k_transpose_to_bf16(const float* __restrict__ W,
                                    __bf16* __restrict__ WT, int K, int N) {
  int n = blockIdx.x * 16 + threadIdx.x;
  int k = blockIdx.y * 16 + threadIdx.y;
  if (n < N && k < K)
    WT[(size_t)n * K + k] = (__bf16)W[(size_t)k * N + n];
}

__global__ void k_zero_u32(unsigned* p) { *p = 0u; }

// ---------------- GI = X @ Wi + bi  (M=T*B=32768, N=3H=1536, K=512) ----------------
// Massively parallel (49152 waves): latency hidden by occupancy; just
// double-buffer the A fragment so B loads overlap WMMA.
__global__ void k_gemm_gi(const __bf16* __restrict__ Xbf,   // [M x K]
                          const __bf16* __restrict__ WiT,   // [N x K]
                          const float*  __restrict__ bi,    // [N]
                          float* __restrict__ GI)           // [M x N]
{
  const int lane = threadIdx.x & 31;
  const int wave = threadIdx.x >> 5;
  const int half = lane >> 4;
  const int lcol = lane & 15;
  const int row0 = blockIdx.y * 64 + wave * 16;
  const int n0   = blockIdx.x * 64;

  v8f acc[4] = {};
  const __bf16* aBase = Xbf + (size_t)(row0 + lcol) * DIM;

  v8bf alo = *(const v8bf*)(aBase + half * 16);
  v8bf ahi = *(const v8bf*)(aBase + 8 + half * 16);

  for (int k0 = 0; k0 < DIM; k0 += 32) {
    v16bf a = combine_frag(alo, ahi);
    if (k0 + 32 < DIM) {  // prefetch next A chunk while WMMAs run
      alo = *(const v8bf*)(aBase + (k0 + 32) + half * 16);
      ahi = *(const v8bf*)(aBase + (k0 + 40) + half * 16);
    }
    __builtin_prefetch(WiT + (size_t)(n0 + lcol) * DIM + k0 + 64, 0, 3);
#pragma unroll
    for (int nt = 0; nt < 4; ++nt) {
      v16bf b = *(const v16bf*)(WiT + (size_t)(n0 + nt * 16 + lcol) * DIM + k0 + half * 16);
      acc[nt] = __builtin_amdgcn_wmma_f32_16x16x32_bf16(
          false, a, false, b, (short)0, acc[nt], false, false);
    }
  }

  // C layout: lane L, vgpr v -> m = v + 8*half, n = lcol
#pragma unroll
  for (int nt = 0; nt < 4; ++nt) {
#pragma unroll
    for (int v = 0; v < 8; ++v) {
      int m = row0 + v + 8 * half;
      int n = n0 + nt * 16 + lcol;
      GI[(size_t)m * H3 + n] = acc[nt][v] + bi[n];
    }
  }
}

// ---------------- persistent recurrent scan ----------------
// grid = 32 blocks (16 hidden columns each), block = 128 threads (4 waves).
// wave w: rows w*16..w*16+15 of B=64; 3 gate tiles share the same column
// slice. Wh slice (48KB) async-staged into LDS ONCE, reused for all 512
// steps. Device-wide barrier (monotonic atomic counter) between steps.
__global__ void __launch_bounds__(128)
k_gru_scan(__bf16* __restrict__ hA,          // [B x H] bf16 (holds h0 at entry)
           __bf16* __restrict__ hB,          // [B x H] bf16 scratch
           const __bf16* __restrict__ WhT,   // [3H x H] bf16
           const float*  __restrict__ GI,    // [T*B x 3H]
           const float*  __restrict__ bhn,   // [H]
           float* __restrict__ out,          // [T x B x H]
           unsigned* __restrict__ barrier_ctr)
{
  __shared__ __attribute__((aligned(32))) __bf16 sW[3 * 16 * HID];  // 48 KB

  const int tid  = threadIdx.x;
  const int lane = tid & 31;
  const int wave = tid >> 5;
  const int half = lane >> 4;
  const int lcol = lane & 15;
  const int row0 = wave * 16;
  const int j0   = blockIdx.x * 16;

  // ---- async stage Wh slice ONCE: 3 segments of 16 contiguous rows ----
#pragma unroll
  for (int g = 0; g < 3; ++g) {
    const __bf16* src = WhT + (size_t)(g * HID + j0) * HID;  // 16 rows x 512 contiguous
#pragma unroll
    for (int it = 0; it < 8; ++it) {                         // 8 x (128 lanes x 16B) = 16KB
      int e = (it * 128 + tid) * 8;
      async_copy_b128(src + e, &sW[g * 16 * HID + e]);
    }
  }
  wait_async_zero();
  __syncthreads();

  const int   j  = j0 + lcol;
  const float bh = bhn[j];
  const unsigned nblocks = gridDim.x;

#pragma unroll 1
  for (int t = 0; t < T_STEPS; ++t) {
    const __bf16* hin  = (t & 1) ? hB : hA;
    __bf16*       hout = (t & 1) ? hA : hB;

    // ---- preload epilogue operands; overlaps the GEMM below ----
    float giR[8], giZ[8], giN[8], hp[8];
#pragma unroll
    for (int v = 0; v < 8; ++v) {
      int b = row0 + v + 8 * half;
      const float* gi = GI + ((size_t)t * BATCH + b) * H3;
      giR[v] = gi[j];
      giZ[v] = gi[512 + j];
      giN[v] = gi[1024 + j];
      hp[v]  = (float)hin[(size_t)b * HID + j];
    }

    // ---- K loop: A from registers (double-buffered global), B from LDS ----
    const __bf16* aBase = hin + (size_t)(row0 + lcol) * HID;
    v8bf alo = *(const v8bf*)(aBase + half * 16);
    v8bf ahi = *(const v8bf*)(aBase + 8 + half * 16);

    v8f accR = {}, accZ = {}, accN = {};
#pragma unroll 4
    for (int k0 = 0; k0 < HID; k0 += 32) {
      v16bf a = combine_frag(alo, ahi);
      if (k0 + 32 < HID) {
        alo = *(const v8bf*)(aBase + (k0 + 32) + half * 16);
        ahi = *(const v8bf*)(aBase + (k0 + 40) + half * 16);
      }
      const int bbase = lcol * HID + k0 + half * 16;
      v16bf br = *(const v16bf*)(&sW[0 * 16 * HID + bbase]);
      v16bf bz = *(const v16bf*)(&sW[1 * 16 * HID + bbase]);
      v16bf bn = *(const v16bf*)(&sW[2 * 16 * HID + bbase]);
      accR = __builtin_amdgcn_wmma_f32_16x16x32_bf16(false, a, false, br, (short)0, accR, false, false);
      accZ = __builtin_amdgcn_wmma_f32_16x16x32_bf16(false, a, false, bz, (short)0, accZ, false, false);
      accN = __builtin_amdgcn_wmma_f32_16x16x32_bf16(false, a, false, bn, (short)0, accN, false, false);
    }

    // ---- fused gate epilogue ----
#pragma unroll
    for (int v = 0; v < 8; ++v) {
      int b = row0 + v + 8 * half;
      float r  = sigmoidf_(giR[v] + accR[v]);
      float z  = sigmoidf_(giZ[v] + accZ[v]);
      float n  = tanh_fast(giN[v] + r * (accN[v] + bh));
      float hn = (1.0f - z) * n + z * hp[v];
      out[((size_t)t * BATCH + b) * HID + j] = hn;
      hout[(size_t)b * HID + j] = (__bf16)hn;
    }

    // ---- device-wide barrier: monotonic counter, release/acquire ----
    __syncthreads();                       // block's stores issued
    if (tid == 0) {
      __threadfence();                     // make hout visible device-wide
      __hip_atomic_fetch_add(barrier_ctr, 1u, __ATOMIC_RELEASE,
                             __HIP_MEMORY_SCOPE_AGENT);
      const unsigned target = (unsigned)(t + 1) * nblocks;
      while (__hip_atomic_load(barrier_ctr, __ATOMIC_ACQUIRE,
                               __HIP_MEMORY_SCOPE_AGENT) < target) {
        __builtin_amdgcn_s_sleep(2);
      }
    }
    __syncthreads();                       // rest of block waits on lane 0
  }
}

// ---------------- host ----------------

extern "C" void kernel_launch(void* const* d_in, const int* in_sizes, int n_in,
                              void* d_out, int out_size, void* d_ws, size_t ws_size,
                              hipStream_t stream) {
  const float* x   = (const float*)d_in[0];  // [T,B,D]
  const float* h0  = (const float*)d_in[1];  // [B,H]
  const float* Wi  = (const float*)d_in[2];  // [D,3H]
  const float* Wh  = (const float*)d_in[3];  // [H,3H]
  const float* bi  = (const float*)d_in[4];  // [3H]
  const float* bhn = (const float*)d_in[5];  // [H]
  float* out = (float*)d_out;

  char* ws = (char*)d_ws;
  size_t off = 0;
  auto alloc = [&](size_t bytes) -> void* {
    void* p = ws + off;
    off += (bytes + 255) & ~(size_t)255;
    return p;
  };
  __bf16*   WiT = (__bf16*)alloc((size_t)H3 * DIM * sizeof(__bf16));
  __bf16*   WhT = (__bf16*)alloc((size_t)H3 * HID * sizeof(__bf16));
  __bf16*   Xbf = (__bf16*)alloc((size_t)T_STEPS * BATCH * DIM * sizeof(__bf16));
  __bf16*   hA  = (__bf16*)alloc((size_t)BATCH * HID * sizeof(__bf16));
  __bf16*   hB  = (__bf16*)alloc((size_t)BATCH * HID * sizeof(__bf16));
  unsigned* ctr = (unsigned*)alloc(sizeof(unsigned));
  float*    GI  = (float*) alloc((size_t)T_STEPS * BATCH * H3 * sizeof(float));
  (void)ws_size;

  // one-time conversions + barrier counter reset
  k_zero_u32<<<1, 1, 0, stream>>>(ctr);
  {
    int n = T_STEPS * BATCH * DIM;
    k_f32_to_bf16<<<(n + 255) / 256, 256, 0, stream>>>(x, Xbf, n);
  }
  {
    int n = BATCH * HID;
    k_f32_to_bf16<<<(n + 255) / 256, 256, 0, stream>>>(h0, hA, n);
  }
  {
    dim3 g(H3 / 16, DIM / 16), b(16, 16);
    k_transpose_to_bf16<<<g, b, 0, stream>>>(Wi, WiT, DIM, H3);
    k_transpose_to_bf16<<<g, b, 0, stream>>>(Wh, WhT, HID, H3);
  }

  // GI = X @ Wi + bi  (all timesteps, no recurrence)
  {
    dim3 g(H3 / 64, (T_STEPS * BATCH) / 64);  // 24 x 512 blocks
    k_gemm_gi<<<g, 128, 0, stream>>>(Xbf, WiT, bi, GI);
  }

  // one persistent launch runs the whole 512-step scan
  k_gru_scan<<<NBLK, 128, 0, stream>>>(hA, hB, WhT, GI, bhn, out, ctr);
}
